// Multi_head_2216203124964
// MI455X (gfx1250) — matrix operand order repaired and checked by reading, not compiled, not run on hardware
//
#include <hip/hip_runtime.h>

// ---------------------------------------------------------------------------
// Multi-head time-gated MoE MLP for MI455X (gfx1250, wave32, WMMA + TDM).
//
// Roofline: 2.2 GFLOP useful math vs 69 MB of HBM traffic -> ~750 TFLOPS
// needed to stay memory bound at 23.3 TB/s. Only the 16x16x32 f16 WMMA path
// gets there; fp32 accuracy is recovered with an f16 hi/lo split (3 WMMAs
// per logical fp32 GEMM step), accumulating in f32.
// Weight staging to LDS uses the Tensor Data Mover (one 16 KB descriptor per
// expert, TENSORcnt-tracked), with a plain VGPR-copy fallback if the TDM
// builtin is unavailable.
// ---------------------------------------------------------------------------

typedef __attribute__((ext_vector_type(16))) _Float16     v16h;
typedef __attribute__((ext_vector_type(8)))  float        v8f;
typedef __attribute__((ext_vector_type(4)))  unsigned int u32x4;
typedef __attribute__((ext_vector_type(8)))  int          i32x8;
typedef __attribute__((ext_vector_type(4)))  int          i32x4;

#define B_TOTAL     262144
#define NEXP        5
#define FRAG_ELEMS  512               // 32 lanes * 16 halfs per fragment
#define EXP_ELEMS   (8 * FRAG_ELEMS)  // 4 N-tiles * 2 K-chunks = 8 fragments
#define EXP_HALFS   (2 * EXP_ELEMS)   // hi block + lo block, contiguous
#define TOTAL_ELEMS (NEXP * EXP_ELEMS)

#if defined(__HIP_DEVICE_COMPILE__) &&                                         \
    __has_builtin(__builtin_amdgcn_tensor_load_to_lds) &&                      \
    __has_builtin(__builtin_amdgcn_s_wait_tensorcnt)
#define USE_TDM 1
#else
#define USE_TDM 0
#endif

// ---------------------------------------------------------------------------
// Kernel 1: pre-pack W0 (5 x 64 x 64 fp32, layout [e][k][n]) into WMMA
// B-matrix fragments in f16 hi/lo split form.
// B layout used here for V_WMMA_F32_16X16X32_F16 (32Kx16N), contiguous k order matching A:
//   lane l: column n = l%16; halfword j holds K = j + 16*(l>=16).
// ws layout: [e]( hi[nt][kc][lane][j] (4096 halfs) | lo[...] (4096 halfs) )
// so each expert is one contiguous 16 KB block for the TDM.
// ---------------------------------------------------------------------------
__global__ void pack_w0_kernel(const float* __restrict__ W0,
                               _Float16* __restrict__ ws) {
    int idx = blockIdx.x * blockDim.x + threadIdx.x;
    if (idx >= TOTAL_ELEMS) return;
    int j    = idx & 15;
    int lane = (idx >> 4) & 31;
    int frag = idx >> 9;
    int kc   = frag & 1;
    int nt   = (frag >> 1) & 3;
    int e    = frag >> 3;
    int n = nt * 16 + (lane & 15);
    int k = kc * 32 + ((lane >= 16) ? 16 : 0) + j;
    float w = W0[(e * 64 + k) * 64 + n];
    _Float16 hi = (_Float16)w;
    _Float16 lo = (_Float16)(w - (float)hi);
    int local = idx & (EXP_ELEMS - 1);
    ws[e * EXP_HALFS + local]             = hi;
    ws[e * EXP_HALFS + EXP_ELEMS + local] = lo;
}

// ---------------------------------------------------------------------------
// Kernel 2: 8 waves per block, each wave computes a 16-row tile.
// ---------------------------------------------------------------------------
__launch_bounds__(256)
__global__ void moe_wmma_kernel(const float* __restrict__ x,      // [B][65]
                                const _Float16* __restrict__ ws,  // packed W0
                                const float* __restrict__ b0,     // [5][64]
                                const float* __restrict__ tw0,    // [5][64]
                                const float* __restrict__ W1,     // [5][64]
                                const float* __restrict__ b1,     // [5]
                                const float* __restrict__ tw1,    // [5]
                                float* __restrict__ out) {        // [B]
    // hi fragments then lo fragments for the current expert (16 KB)
    __shared__ __attribute__((aligned(32))) _Float16 sB[EXP_HALFS];
    _Float16* sBhi = sB;
    _Float16* sBlo = sB + EXP_ELEMS;

    const int tid  = threadIdx.x;
    const int lane = tid & 31;
    const int wave = tid >> 5;
    const int half = (lane >= 16) ? 1 : 0;   // lane half selects M/K sub-block
    const int ln   = lane & 15;
    const int tile0 = (blockIdx.x * 8 + wave) * 16;

    // --- A fragments: 16x32 f16, hi/lo split, held in registers -----------
    // A layout (16-bit, 16x32): lane = row m (%16); halfword j holds
    //   K = j + 16*(lane>=16)   (contiguous, matching packed B).
    const float* pf = x + (size_t)(tile0 + ln) * 65 + 1;
    const int koff = half * 16;
    v16h ahi[2], alo[2];
#pragma unroll
    for (int kc = 0; kc < 2; ++kc) {
#pragma unroll
        for (int j = 0; j < 16; ++j) {
            int k = kc * 32 + koff + j;
            float v = pf[k];
            _Float16 h = (_Float16)v;
            ahi[kc][j] = h;
            alo[kc][j] = (_Float16)(v - (float)h);
        }
    }

    // --- per-row t and expert index (C layout: vgpr r -> row r + 8*half) --
    float trow[8];
    int   eidx[8];
#pragma unroll
    for (int r = 0; r < 8; ++r) {
        float t = x[(size_t)(tile0 + r + 8 * half) * 65];
        trow[r] = t;
        // searchsorted(BOUNDS, t, side='right') == count of bounds <= t
        eidx[r] = (t >= 0.6f) + (t >= 0.7f) + (t >= 0.8f) + (t >= 0.9f);
    }

    float result[8];
#pragma unroll
    for (int r = 0; r < 8; ++r) result[r] = 0.0f;

    for (int e = 0; e < NEXP; ++e) {
        __syncthreads();   // previous expert's LDS readers are done

#if USE_TDM
        // --- Tensor Data Mover: one 16 KB 1-D descriptor per expert -------
        // D# group0: count=1 | lds_addr | global_addr[56:0] | type=2
        // D# group1: data_size=3 (8B) ; tensor_dim0=tile_dim0=2048 ;
        //            tensor_dim1=tile_dim1=1 ; dim0_stride=2048 ; mask=0
        if (wave == 0) {
            unsigned long long ga =
                (unsigned long long)(size_t)(ws + (size_t)e * EXP_HALFS);
            unsigned lds_addr = (unsigned)(size_t)(void*)sB;  // addr[31:0]
            u32x4 g0;
            g0.x = 1u;                                   // count = 1
            g0.y = lds_addr;                             // D#.lds_addr
            g0.z = (unsigned)(ga & 0xffffffffu);         // global_addr[31:0]
            g0.w = (unsigned)((ga >> 32) & 0x01ffffffu)  // global_addr[56:32]
                   | (2u << 30);                         // type = 2 ("image")
            i32x8 g1;
            g1[0] = (3 << 16);          // data_size = 3 -> 8-byte elements
            g1[1] = (int)(2048u << 16); // tensor_dim0[15:0] (=2048) in [31:16]
            g1[2] = (1 << 16);          // tensor_dim0[31:16]=0 | tensor_dim1=1
            g1[3] = (int)(2048u << 16); // tensor_dim1[31:16]=0 | tile_dim0=2048
            g1[4] = 1;                  // tile_dim1 = 1, tile_dim2 = 0
            g1[5] = 2048;               // tensor_dim0_stride[31:0]
            g1[6] = 0;                  // stride[47:32]=0 | dim1_stride lo
            g1[7] = 0;
            i32x4 gz4 = {0, 0, 0, 0};   // groups 2/3: NULL (<=2-D tensor)
            i32x8 gz8 = {0, 0, 0, 0, 0, 0, 0, 0};  // unused extra group
            __builtin_amdgcn_tensor_load_to_lds(g0, g1, gz4, gz4, gz8, 0);
            __builtin_amdgcn_s_wait_tensorcnt(0);
        }
#else
        // --- fallback: cooperative VGPR copy (2 x v16h per thread) --------
        {
            const v16h* g = (const v16h*)(ws + (size_t)e * EXP_HALFS);
            ((v16h*)sB)[tid]       = g[tid];
            ((v16h*)sB)[tid + 256] = g[tid + 256];
        }
#endif
        __syncthreads();   // staged data visible to all 8 waves

        float partial[8];
#pragma unroll
        for (int r = 0; r < 8; ++r) partial[r] = 0.0f;

#pragma unroll
        for (int nt = 0; nt < 4; ++nt) {
            v8f acc = {};
#pragma unroll
            for (int kc = 0; kc < 2; ++kc) {
                const v16h bh =
                    *(const v16h*)(sBhi + (nt * 2 + kc) * FRAG_ELEMS + lane * 16);
                const v16h bl =
                    *(const v16h*)(sBlo + (nt * 2 + kc) * FRAG_ELEMS + lane * 16);
                // fp32-accurate product via hi/lo split: hh + lh + hl
                acc = __builtin_amdgcn_wmma_f32_16x16x32_f16(
                    false, ahi[kc], false, bh, (short)0, acc, false, false);
                acc = __builtin_amdgcn_wmma_f32_16x16x32_f16(
                    false, alo[kc], false, bh, (short)0, acc, false, false);
                acc = __builtin_amdgcn_wmma_f32_16x16x32_f16(
                    false, ahi[kc], false, bl, (short)0, acc, false, false);
            }
            // layer-1 epilogue + layer-2 partial dot, directly on C layout:
            // lane holds column n = nt*16 + ln; vgpr r holds row r + 8*half.
            int n = nt * 16 + ln;
            float tw0v = tw0[e * 64 + n];
            float b0v  = b0[e * 64 + n];
            float w1v  = W1[e * 64 + n];
#pragma unroll
            for (int r = 0; r < 8; ++r) {
                float h = acc[r] + trow[r] * tw0v + b0v;
                h = fmaxf(h, 0.0f);
                partial[r] = fmaf(h, w1v, partial[r]);
            }
        }

        // Reduce layer-2 dot across the 16 lanes of each half (wave32).
#pragma unroll
        for (int off = 1; off <= 8; off <<= 1) {
#pragma unroll
            for (int r = 0; r < 8; ++r)
                partial[r] += __shfl_xor(partial[r], off, 32);
        }

        float tw1v = tw1[e];
        float b1v  = b1[e];
#pragma unroll
        for (int r = 0; r < 8; ++r) {
            float o = partial[r] + trow[r] * tw1v + b1v;
            if (eidx[r] == e) result[r] = o;
        }
    }

    // Lane 0 of each half writes its 8 rows.
    if (ln == 0) {
#pragma unroll
        for (int r = 0; r < 8; ++r)
            out[tile0 + r + 8 * half] = result[r];
    }
}

extern "C" void kernel_launch(void* const* d_in, const int* in_sizes, int n_in,
                              void* d_out, int out_size, void* d_ws, size_t ws_size,
                              hipStream_t stream) {
    const float* x   = (const float*)d_in[0];
    const float* W0  = (const float*)d_in[1];
    const float* b0  = (const float*)d_in[2];
    const float* tw0 = (const float*)d_in[3];
    const float* W1  = (const float*)d_in[4];
    const float* b1  = (const float*)d_in[5];
    const float* tw1 = (const float*)d_in[6];
    float* out = (float*)d_out;

    _Float16* ws = (_Float16*)d_ws;   // 80 KB packed W0 fragments

    pack_w0_kernel<<<(TOTAL_ELEMS + 255) / 256, 256, 0, stream>>>(W0, ws);
    moe_wmma_kernel<<<B_TOTAL / 128, 256, 0, stream>>>(
        x, ws, b0, tw0, W1, b1, tw1, out);
}
